// MultiHeadAttention_13383118094532
// MI455X (gfx1250) — compile-verified, matrix operand order checked by reading, not run on hardware
//
#include <hip/hip_runtime.h>
#include <hip/hip_bf16.h>

// CDNA5 / gfx1250 fused tiny-transformer forward + CE loss.
// One wave32 per block handles 16 tokens (2 batch rows, T=8).
// All dense layers use v_wmma_f32_16x16x32_f16 (K=E=32 per step).
// Weights are pre-swizzled once into WMMA B-fragment layout (f16) in d_ws.

typedef __attribute__((ext_vector_type(16))) _Float16 v16h;
typedef __attribute__((ext_vector_type(8)))  float    v8f;

#define E_DIM 32
#define T_BLK 8
#define HEADS 4
#define HS 8
#define HID 128
#define VOC 65
#define VPAD 80
#define NLAYER 3

#define TILES_PER_LAYER 24     // qkv:6, proj:2, w1:8, w2:8
#define NTILES (NLAYER * TILES_PER_LAYER + 5)   // + lm_head:5  => 77
#define WPACK_BYTES (80 * 1024)                 // packed-weight region in d_ws

struct __align__(16) Smem {
    float    x0[16][E_DIM];   // token+pos embedding (kept for final residual)
    float    h [16][E_DIM];   // running residual stream
    float    qb[16][E_DIM];   // q (heads concatenated)
    float    kb[16][E_DIM];   // k
    float    vb[16][E_DIM];   // v
    float    lb[16][VPAD];    // logits scratch (65 valid, padded to 80)
    _Float16 ast[16][E_DIM];  // f16 A-stage (xn / att / xn2 / final x)
    _Float16 hst[16][HID];    // f16 MLP hidden stage
};

__device__ __forceinline__ float wave_sum32(float x) {
#pragma unroll
    for (int m = 16; m >= 1; m >>= 1) x += __shfl_xor(x, m, 32);
    return x;
}

__device__ __forceinline__ v8f wmma16(v16h a, v16h b, v8f c) {
    // D = A(16x32 f16) * B(32x16 f16) + C(16x16 f32)
    return __builtin_amdgcn_wmma_f32_16x16x32_f16(
        false, a, false, b, (short)0, c, false, false);
}

// ---- operand gather per ISA 7.12.2 16-bit layouts (wave32) ------------------
// A 16x32 (MxK): lane L holds row m=L&15; kbase=(L>>4)*8; elem i -> k = kbase + (i<8 ? i : i+8)
__device__ __forceinline__ v16h pack_A(const _Float16* a, int ldk, int k0) {
    const int lane = threadIdx.x & 31;
    const int m  = lane & 15;
    const int kb = ((lane >> 4) << 3) + k0;
    v16h r;
#pragma unroll
    for (int i = 0; i < 16; ++i) {
        const int k = kb + (i < 8 ? i : i + 8);
        r[i] = a[m * ldk + k];
    }
    return r;
}

// B 32x16 (KxN) gathered from row-major f32 weights [K][ldn]  (prep kernel only)
__device__ __forceinline__ v16h gather_B_rm(const float* w, int ldn, int n0, int k0, int ncols) {
    const int lane = threadIdx.x & 31;
    const int n  = n0 + (lane & 15);
    const int kb = ((lane >> 4) << 3) + k0;
    v16h r;
#pragma unroll
    for (int i = 0; i < 16; ++i) {
        const int k = kb + (i < 8 ? i : i + 8);
        const float v = (n < ncols) ? w[k * ldn + n] : 0.0f;
        r[i] = (_Float16)v;
    }
    return r;
}

// B for QKV: column n = h*HS+d maps into wq[l] layout (H,E,HS)  (prep kernel only)
__device__ __forceinline__ v16h gather_B_qkv(const float* wl, int n0) {
    const int lane = threadIdx.x & 31;
    const int n  = n0 + (lane & 15);
    const int hh = n >> 3, d = n & 7;
    const int kb = (lane >> 4) << 3;
    v16h r;
#pragma unroll
    for (int i = 0; i < 16; ++i) {
        const int e = kb + (i < 8 ? i : i + 8);
        r[i] = (_Float16)wl[(hh * E_DIM + e) * HS + d];
    }
    return r;
}

// main kernel: fetch a pre-swizzled B fragment (32 contiguous bytes per lane)
__device__ __forceinline__ v16h load_frag(const _Float16* wp, int tile) {
    const int lane = threadIdx.x & 31;
    return *reinterpret_cast<const v16h*>(wp + (((size_t)tile) << 9) + (lane << 4));
}

// C/D 16x16 f32: elem r -> (m = r + (lane>=16 ? 8:0), n = lane&15)
__device__ __forceinline__ v8f make_C_bias(const float* bias, int n0, int ncols) {
    const int lane = threadIdx.x & 31;
    const int n = n0 + (lane & 15);
    const float bv = (n < ncols) ? bias[n] : 0.0f;
    v8f c;
#pragma unroll
    for (int r = 0; r < 8; ++r) c[r] = bv;
    return c;
}

__device__ __forceinline__ v8f make_C_res_bias(const float (*hb)[E_DIM], const float* bias, int n0) {
    const int lane = threadIdx.x & 31;
    const int n  = n0 + (lane & 15);
    const int mb = (lane >> 4) << 3;
    const float bv = bias[n];
    v8f c;
#pragma unroll
    for (int r = 0; r < 8; ++r) c[r] = hb[r + mb][n] + bv;
    return c;
}

__device__ __forceinline__ void store_D(float* out, int ldn, int n0, v8f d) {
    const int lane = threadIdx.x & 31;
    const int n  = n0 + (lane & 15);
    const int mb = (lane >> 4) << 3;
#pragma unroll
    for (int r = 0; r < 8; ++r) out[(r + mb) * ldn + n] = d[r];
}

__device__ __forceinline__ void store_D_relu_f16(_Float16* out, int ldn, int n0, v8f d) {
    const int lane = threadIdx.x & 31;
    const int n  = n0 + (lane & 15);
    const int mb = (lane >> 4) << 3;
#pragma unroll
    for (int r = 0; r < 8; ++r) {
        float v = d[r];
        out[(r + mb) * ldn + n] = (_Float16)(v > 0.0f ? v : 0.0f);
    }
}

// layernorm over E=32 (one lane per channel), optional extra residual add, f16 out
__device__ __forceinline__ void layernorm32(const float (*in)[E_DIM],
                                            const float* g, const float* b,
                                            const float (*addres)[E_DIM],
                                            _Float16 (*out)[E_DIM], int lane) {
    const float gv = g[lane], bv = b[lane];
#pragma unroll
    for (int r = 0; r < 16; ++r) {
        const float x = in[r][lane];
        const float mean = wave_sum32(x) * (1.0f / 32.0f);
        const float d = x - mean;
        const float var = wave_sum32(d * d) * (1.0f / 32.0f);
        float y = d * rsqrtf(var + 1e-5f) * gv + bv;
        if (addres) y += addres[r][lane];
        out[r][lane] = (_Float16)y;
    }
}

// causal attention: 2 batch rows x 8 tokens, 4 heads of 8; lane = (row, head-pair)
__device__ __forceinline__ void attention8(const float (*qb)[E_DIM], const float (*kb)[E_DIM],
                                           const float (*vb)[E_DIM], _Float16 (*out)[E_DIM],
                                           int lane) {
    const int r  = lane & 15;
    const int rb = r & 8;          // batch-local base row
    const int t  = r & 7;          // token index
    const int h0 = lane >> 4;      // head group 0/1
    const float scale = 0.17677669529663689f;  // E^-0.5 = 1/sqrt(32)
#pragma unroll
    for (int hi = 0; hi < 2; ++hi) {
        const int cb = (h0 + 2 * hi) * HS;     // head column base
        float w[8];
        float mx = -3.0e38f;
#pragma unroll
        for (int s = 0; s < 8; ++s) {
            float dot = 0.0f;
#pragma unroll
            for (int d = 0; d < 8; ++d) dot += qb[r][cb + d] * kb[rb + s][cb + d];
            w[s] = (s <= t) ? dot * scale : -1.0e9f;
            mx = fmaxf(mx, w[s]);
        }
        float se = 0.0f;
#pragma unroll
        for (int s = 0; s < 8; ++s) { w[s] = __expf(w[s] - mx); se += w[s]; }
        const float inv = 1.0f / se;
#pragma unroll
        for (int d = 0; d < 8; ++d) {
            float acc = 0.0f;
#pragma unroll
            for (int s = 0; s < 8; ++s) acc += w[s] * vb[rb + s][cb + d];
            out[r][cb + d] = (_Float16)(acc * inv);
        }
    }
}

// one-shot: swizzle all weight matrices into WMMA B-fragment layout (f16)
__global__ __launch_bounds__(32) void pack_weights_kernel(
    const float* __restrict__ wq, const float* __restrict__ wk, const float* __restrict__ wv,
    const float* __restrict__ proj_w, const float* __restrict__ w1, const float* __restrict__ w2,
    const float* __restrict__ lm_w, _Float16* __restrict__ out)
{
    const int tile = blockIdx.x;
    const int lane = threadIdx.x & 31;
    v16h frag;
    if (tile < NLAYER * TILES_PER_LAYER) {
        const int l = tile / TILES_PER_LAYER;
        const int t = tile % TILES_PER_LAYER;
        if (t < 6) {                       // wq, wk, wv : 2 tiles each
            const float* base = (t < 2) ? wq : (t < 4) ? wk : wv;
            frag = gather_B_qkv(base + l * (HEADS * E_DIM * HS), (t & 1) * 16);
        } else if (t < 8) {                // proj_w : 2 tiles
            frag = gather_B_rm(proj_w + l * E_DIM * E_DIM, E_DIM, (t - 6) * 16, 0, E_DIM);
        } else if (t < 16) {               // w1 : 8 N-tiles
            frag = gather_B_rm(w1 + l * E_DIM * HID, HID, (t - 8) * 16, 0, HID);
        } else {                           // w2 : 2 N-tiles x 4 K-tiles
            const int nt = (t - 16) >> 2, kt = (t - 16) & 3;
            frag = gather_B_rm(w2 + l * HID * E_DIM, E_DIM, nt * 16, kt * 32, E_DIM);
        }
    } else {                               // lm_w : 5 N-tiles (65 cols padded)
        frag = gather_B_rm(lm_w, VOC, (tile - NLAYER * TILES_PER_LAYER) * 16, 0, VOC);
    }
    *reinterpret_cast<v16h*>(out + (((size_t)tile) << 9) + (lane << 4)) = frag;
}

__global__ __launch_bounds__(32) void gpt_forward_kernel(
    const int* __restrict__ idx, const int* __restrict__ targets,
    const float* __restrict__ tok_emb, const float* __restrict__ pos_emb,
    const float* __restrict__ proj_b,
    const float* __restrict__ ln1_g, const float* __restrict__ ln1_b,
    const float* __restrict__ ln2_g, const float* __restrict__ ln2_b,
    const float* __restrict__ b1, const float* __restrict__ b2,
    const float* __restrict__ lnf_g, const float* __restrict__ lnf_b,
    const float* __restrict__ lm_b,
    const _Float16* __restrict__ wp,        // packed B fragments
    float* __restrict__ logits_out, float* __restrict__ loss_part)
{
    __shared__ Smem sm;
    const int lane = threadIdx.x & 31;
    const int row0 = blockIdx.x * 16;   // 16 tokens = 2 batch rows

    // embeddings: x0 = tok_emb[idx] + pos_emb[t]
#pragma unroll
    for (int r = 0; r < 16; ++r) {
        const int R = row0 + r;
        const int tok = idx[R];
        const float v = tok_emb[tok * E_DIM + lane] + pos_emb[(R & 7) * E_DIM + lane];
        sm.x0[r][lane] = v;
        sm.h[r][lane]  = v;
    }
    __syncthreads();

    for (int l = 0; l < NLAYER; ++l) {
        const int tb = l * TILES_PER_LAYER;

        // ---- LN1 -> ast ----
        layernorm32(sm.h, ln1_g + l * E_DIM, ln1_b + l * E_DIM,
                    (const float(*)[E_DIM])nullptr, sm.ast, lane);
        __syncthreads();

        // ---- QKV (K=32, two 16-col tiles each) ----
        {
            const v16h a = pack_A(&sm.ast[0][0], E_DIM, 0);
#pragma unroll
            for (int nt = 0; nt < 2; ++nt) {
                v8f c = {};
                c = wmma16(a, load_frag(wp, tb + 0 + nt), c);
                store_D(&sm.qb[0][0], E_DIM, nt * 16, c);
            }
#pragma unroll
            for (int nt = 0; nt < 2; ++nt) {
                v8f c = {};
                c = wmma16(a, load_frag(wp, tb + 2 + nt), c);
                store_D(&sm.kb[0][0], E_DIM, nt * 16, c);
            }
#pragma unroll
            for (int nt = 0; nt < 2; ++nt) {
                v8f c = {};
                c = wmma16(a, load_frag(wp, tb + 4 + nt), c);
                store_D(&sm.vb[0][0], E_DIM, nt * 16, c);
            }
        }
        __syncthreads();

        // ---- causal softmax attention (VALU; 8x8 blocks too small for WMMA) ----
        attention8(sm.qb, sm.kb, sm.vb, sm.ast, lane);
        __syncthreads();

        // ---- output proj + residual: h = h + att @ proj_w + proj_b ----
        {
            const v16h a = pack_A(&sm.ast[0][0], E_DIM, 0);
            const float* pb = proj_b + l * E_DIM;
            v8f c0 = make_C_res_bias(sm.h, pb, 0);
            v8f c1 = make_C_res_bias(sm.h, pb, 16);
            c0 = wmma16(a, load_frag(wp, tb + 6), c0);
            c1 = wmma16(a, load_frag(wp, tb + 7), c1);
            store_D(&sm.h[0][0], E_DIM, 0,  c0);
            store_D(&sm.h[0][0], E_DIM, 16, c1);
        }
        __syncthreads();

        // ---- LN2 -> ast ----
        layernorm32(sm.h, ln2_g + l * E_DIM, ln2_b + l * E_DIM,
                    (const float(*)[E_DIM])nullptr, sm.ast, lane);
        __syncthreads();

        // ---- MLP up: hid = relu(xn2 @ w1 + b1), 8 N-tiles, f16 out ----
        {
            const v16h a = pack_A(&sm.ast[0][0], E_DIM, 0);
            const float* b1l = b1 + l * HID;
#pragma unroll
            for (int nt = 0; nt < 8; ++nt) {
                v8f c = make_C_bias(b1l, nt * 16, HID);
                c = wmma16(a, load_frag(wp, tb + 8 + nt), c);
                store_D_relu_f16(&sm.hst[0][0], HID, nt * 16, c);
            }
        }
        __syncthreads();

        // ---- MLP down + residual: h = h + hid @ w2 + b2 (K=128 -> 4 K-tiles) ----
        {
            const float* b2l = b2 + l * E_DIM;
#pragma unroll
            for (int nt = 0; nt < 2; ++nt) {
                v8f c = make_C_res_bias(sm.h, b2l, nt * 16);
#pragma unroll
                for (int kt = 0; kt < 4; ++kt) {
                    const v16h a = pack_A(&sm.hst[0][0] + kt * 32, HID, 0);
                    c = wmma16(a, load_frag(wp, tb + 16 + nt * 4 + kt), c);
                }
                store_D(&sm.h[0][0], E_DIM, nt * 16, c);
            }
        }
        __syncthreads();
    }

    // ---- final: x = x0 + ln_f(h) -> ast (f16) ----
    layernorm32(sm.h, lnf_g, lnf_b, sm.x0, sm.ast, lane);
    __syncthreads();

    // ---- LM head: logits = x @ lm_w + lm_b, V=65 padded to 5 tiles of 16 ----
    {
        const v16h a = pack_A(&sm.ast[0][0], E_DIM, 0);
#pragma unroll
        for (int nt = 0; nt < 5; ++nt) {
            v8f c = make_C_bias(lm_b, nt * 16, VOC);
            c = wmma16(a, load_frag(wp, NLAYER * TILES_PER_LAYER + nt), c);
            store_D(&sm.lb[0][0], VPAD, nt * 16, c);
        }
    }
    __syncthreads();

    // ---- stream out logits, non-temporal (output > L2, don't thrash it) ----
#pragma unroll
    for (int r = 0; r < 16; ++r) {
        const size_t base = (size_t)(row0 + r) * VOC;
        for (int c = lane; c < VOC; c += 32)
            __builtin_nontemporal_store(sm.lb[r][c], &logits_out[base + c]);
    }

    // ---- per-row CE loss, lanes 0..15 each own a row ----
    float li = 0.0f;
    if (lane < 16) {
        const int r = lane;
        float mx = -3.0e38f;
        for (int c = 0; c < VOC; ++c) mx = fmaxf(mx, sm.lb[r][c]);
        float se = 0.0f;
        for (int c = 0; c < VOC; ++c) se += __expf(sm.lb[r][c] - mx);
        const float logZ = mx + __logf(se);
        const int tgt = targets[row0 + r];
        li = logZ - sm.lb[r][tgt];   // -logp[target]
    }
    const float tot = wave_sum32(li);
    if (lane == 0) loss_part[blockIdx.x] = tot;
}

// deterministic single-block tree reduction of per-block partial losses
__global__ __launch_bounds__(256) void loss_reduce_kernel(const float* __restrict__ part,
                                                          int n, float scale,
                                                          float* __restrict__ out) {
    __shared__ float s[256];
    float acc = 0.0f;
    for (int i = threadIdx.x; i < n; i += 256) acc += part[i];
    s[threadIdx.x] = acc;
    __syncthreads();
    for (int w = 128; w > 0; w >>= 1) {
        if ((int)threadIdx.x < w) s[threadIdx.x] += s[threadIdx.x + w];
        __syncthreads();
    }
    if (threadIdx.x == 0) out[0] = s[0] * scale;
}

extern "C" void kernel_launch(void* const* d_in, const int* in_sizes, int n_in,
                              void* d_out, int out_size, void* d_ws, size_t ws_size,
                              hipStream_t stream) {
    const int*   idx     = (const int*)d_in[0];
    const int*   targets = (const int*)d_in[1];
    const float* tok_emb = (const float*)d_in[2];
    const float* pos_emb = (const float*)d_in[3];
    const float* wq      = (const float*)d_in[4];
    const float* wk      = (const float*)d_in[5];
    const float* wv      = (const float*)d_in[6];
    const float* proj_w  = (const float*)d_in[7];
    const float* proj_b  = (const float*)d_in[8];
    const float* ln1_g   = (const float*)d_in[9];
    const float* ln1_b   = (const float*)d_in[10];
    const float* ln2_g   = (const float*)d_in[11];
    const float* ln2_b   = (const float*)d_in[12];
    const float* w1      = (const float*)d_in[13];
    const float* b1      = (const float*)d_in[14];
    const float* w2      = (const float*)d_in[15];
    const float* b2      = (const float*)d_in[16];
    const float* lnf_g   = (const float*)d_in[17];
    const float* lnf_b   = (const float*)d_in[18];
    const float* lm_w    = (const float*)d_in[19];
    const float* lm_b    = (const float*)d_in[20];

    float* out = (float*)d_out;
    const int nrows = in_sizes[0];       // B*T tokens
    const int nblk  = nrows / 16;        // 16 tokens per block

    _Float16* wpack = (_Float16*)d_ws;                       // 77 KB packed fragments
    float*    part  = (float*)((char*)d_ws + WPACK_BYTES);   // per-block loss partials

    pack_weights_kernel<<<NTILES, 32, 0, stream>>>(wq, wk, wv, proj_w, w1, w2, lm_w, wpack);

    gpt_forward_kernel<<<nblk, 32, 0, stream>>>(
        idx, targets, tok_emb, pos_emb, proj_b,
        ln1_g, ln1_b, ln2_g, ln2_b, b1, b2, lnf_g, lnf_b, lm_b,
        wpack, out, part);

    // loss scalar lives after the flattened logits
    loss_reduce_kernel<<<1, 256, 0, stream>>>(
        part, nblk, 1.0f / (float)nrows, out + ((size_t)out_size - 1));
}
